// QuantLlamaMLP_73349451481849
// MI455X (gfx1250) — matrix operand order repaired and checked by reading, not compiled, not run on hardware
//
#include <hip/hip_runtime.h>
#include <hip/hip_bf16.h>

typedef __attribute__((ext_vector_type(16))) _Float16 v16h;
typedef __attribute__((ext_vector_type(8)))  float    v8f;
typedef __attribute__((ext_vector_type(4)))  int      v4i;

#define D_MODEL 4096
#define D_FF    11008
#define M_DIM   8192
#define GROUPSZ 128

#define KB 64
#define KP 72   // padded LDS row stride (halves): multiple of 8 for 16B-aligned b128

#if __has_builtin(__builtin_amdgcn_global_load_async_to_lds_b128)
#define HAVE_ASYNC_LDS 1
#else
#define HAVE_ASYNC_LDS 0
#endif

#if HAVE_ASYNC_LDS
__device__ __forceinline__ void async_copy16(const void* g, void* l) {
  __builtin_amdgcn_global_load_async_to_lds_b128(
      (__attribute__((address_space(1))) v4i*)g,
      (__attribute__((address_space(3))) v4i*)l, 0, 0);
}
#endif

__device__ __forceinline__ void wait_async0() {
#if __has_builtin(__builtin_amdgcn_s_wait_asynccnt)
  __builtin_amdgcn_s_wait_asynccnt(0);
#else
  asm volatile("s_wait_asynccnt 0x0" ::: "memory");
#endif
}

// ---------------------------------------------------------------------------
// x (f32) -> f16, vectorized by 4
// ---------------------------------------------------------------------------
struct H4 { _Float16 a, b, c, d; };

__global__ __launch_bounds__(256) void awq_x_to_h(const float* __restrict__ x,
                                                  _Float16* __restrict__ xh, int n4) {
  int i = blockIdx.x * 256 + threadIdx.x;
  if (i < n4) {
    const float4 v = ((const float4*)x)[i];
    H4 o{(_Float16)v.x, (_Float16)v.y, (_Float16)v.z, (_Float16)v.w};
    *(H4*)(xh + (size_t)i * 4) = o;
  }
}

// ---------------------------------------------------------------------------
// Dequant q[K][N] (int32 nibble values) with per-group z,s -> wT[N][K] f16
// ---------------------------------------------------------------------------
__global__ __launch_bounds__(256) void awq_dequant_t(const int* __restrict__ q,
                                                     const int* __restrict__ z,
                                                     const float* __restrict__ s,
                                                     _Float16* __restrict__ wT,
                                                     int K, int N) {
  __shared__ __align__(16) _Float16 T[64][72];
  __shared__ float zf[64];
  __shared__ float sf[64];
  const int t  = threadIdx.x;
  const int k0 = blockIdx.y * 64;
  const int n0 = blockIdx.x * 64;
  const int g  = k0 >> 7;  // group index (GROUPSZ==128)
  if (t < 64) {
    zf[t] = (float)z[(size_t)g * N + n0 + t];
    sf[t] = s[(size_t)g * N + n0 + t];
  }
  __syncthreads();
#pragma unroll
  for (int i = 0; i < 16; ++i) {
    int idx = t + i * 256;
    int k = idx >> 6, n = idx & 63;
    float w = ((float)q[(size_t)(k0 + k) * N + n0 + n] - zf[n]) * sf[n];
    T[n][k] = (_Float16)w;
  }
  __syncthreads();
#pragma unroll
  for (int i = 0; i < 2; ++i) {
    int idx = t + i * 256;
    int n = idx >> 3, c = (idx & 7) * 8;
    *(uint4*)&wT[(size_t)(n0 + n) * K + k0 + c] = *(const uint4*)&T[n][c];
  }
}

// ---------------------------------------------------------------------------
// Fragment helpers (layouts per CDNA5 ISA §7.12.2)
// ---------------------------------------------------------------------------
__device__ __forceinline__ v16h load_a_frag(const _Float16* base, int kb, int half) {
  v16h a;
  uint4* p = (uint4*)&a;
  p[0] = *(const uint4*)(base + kb + half * 8);        // K = half*8 + 0..7
  p[1] = *(const uint4*)(base + kb + 16 + half * 8);   // K = 16 + half*8 + 0..7
  return a;
}

__device__ __forceinline__ v16h load_b_frag(const _Float16* base, int kb, int half) {
  v16h b;
  uint4* p = (uint4*)&b;
  p[0] = *(const uint4*)(base + kb + half * 8);        // K = half*8 + 0..7
  p[1] = *(const uint4*)(base + kb + 16 + half * 8);   // K = 16 + half*8 + 0..7
  return b;
}

// ---------------------------------------------------------------------------
// Fused gate/up GEMM + SwiGLU: H[M][D_FF] = silu(x@Wg) * (x@Wu), f16 out.
// BM=256 x BN=128 block, 1024 threads = 32 wave32 (8x4 grid, 32x32 wave
// tiles), KB=64 double-buffered async-LDS pipeline (16 WMMA per barrier).
// Grid: blockIdx.x walks M so concurrent blocks share one N-column's weights.
// ---------------------------------------------------------------------------
#define GBM 256
#define GBN 128

__global__ __launch_bounds__(1024) void awq_gemm_gateup(const _Float16* __restrict__ A,
                                                        const _Float16* __restrict__ Bg,
                                                        const _Float16* __restrict__ Bu,
                                                        _Float16* __restrict__ H) {
  const int Kt = D_MODEL, Nt = D_FF;
  __shared__ __align__(16) _Float16 As[2][GBM][KP];
  __shared__ __align__(16) _Float16 Bgs[2][GBN][KP];
  __shared__ __align__(16) _Float16 Bus[2][GBN][KP];

  const int t    = threadIdx.x;
  const int m0   = blockIdx.x * GBM;   // M fastest -> weight-slab sharing in L2
  const int n0   = blockIdx.y * GBN;
  const int lane = t & 31, w = t >> 5;
  const int wm   = (w >> 2) * 32;  // wave M offset (0..224)
  const int wn   = (w & 3) * 32;   // wave N offset (0..96)
  const int half = lane >> 4, row = lane & 15;
  const int bu_half = t >> 9;      // 0 -> copies Bg, 1 -> copies Bu

  v8f accG[2][2] = {};
  v8f accU[2][2] = {};

  auto compute_stage = [&](int b) {
#pragma unroll
    for (int ks = 0; ks < 2; ++ks) {
      const int kb = ks * 32;
      v16h af[2];
#pragma unroll
      for (int mt = 0; mt < 2; ++mt)
        af[mt] = load_a_frag(&As[b][wm + mt * 16 + row][0], kb, half);
#pragma unroll
      for (int nt = 0; nt < 2; ++nt) {
        const v16h bg = load_b_frag(&Bgs[b][wn + nt * 16 + row][0], kb, half);
        const v16h bu = load_b_frag(&Bus[b][wn + nt * 16 + row][0], kb, half);
#pragma unroll
        for (int mt = 0; mt < 2; ++mt) {
          accG[mt][nt] = __builtin_amdgcn_wmma_f32_16x16x32_f16(
              false, af[mt], false, bg, (short)0, accG[mt][nt], false, false);
          accU[mt][nt] = __builtin_amdgcn_wmma_f32_16x16x32_f16(
              false, af[mt], false, bu, (short)0, accU[mt][nt], false, false);
        }
      }
    }
  };

#if HAVE_ASYNC_LDS
  auto issue_stage = [&](int k0, int b) {
#pragma unroll
    for (int i = 0; i < 2; ++i) {        // A tile: 2048 x 16B, 2 per thread
      int idx = t + i * 1024;
      int r = idx >> 3, c = (idx & 7) * 8;
      async_copy16(A + (size_t)(m0 + r) * Kt + k0 + c, &As[b][r][c]);
    }
#pragma unroll
    for (int i = 0; i < 2; ++i) {        // B tiles: 1024 x 16B each, split halves
      int idx = (t & 511) + i * 512;
      int r = idx >> 3, c = (idx & 7) * 8;
      const _Float16* Bsel = bu_half ? Bu : Bg;
      _Float16* ldst = bu_half ? &Bus[b][r][c] : &Bgs[b][r][c];
      async_copy16(Bsel + (size_t)(n0 + r) * Kt + k0 + c, ldst);
    }
  };
  issue_stage(0, 0);
  for (int k0 = 0; k0 < Kt; k0 += 2 * KB) {  // Kt/KB = 64 stages (even)
    wait_async0();
    __syncthreads();
    if (k0 + KB < Kt) issue_stage(k0 + KB, 1);
    compute_stage(0);
    wait_async0();
    __syncthreads();
    if (k0 + 2 * KB < Kt) issue_stage(k0 + 2 * KB, 0);
    compute_stage(1);
  }
#else
  for (int k0 = 0; k0 < Kt; k0 += KB) {
    __syncthreads();
#pragma unroll
    for (int i = 0; i < 2; ++i) {
      int idx = t + i * 1024;
      int r = idx >> 3, c = (idx & 7) * 8;
      *(uint4*)&As[0][r][c] = *(const uint4*)(A + (size_t)(m0 + r) * Kt + k0 + c);
    }
#pragma unroll
    for (int i = 0; i < 2; ++i) {
      int idx = (t & 511) + i * 512;
      int r = idx >> 3, c = (idx & 7) * 8;
      const _Float16* Bsel = bu_half ? Bu : Bg;
      _Float16* ldst = bu_half ? &Bus[0][r][c] : &Bgs[0][r][c];
      *(uint4*)ldst = *(const uint4*)(Bsel + (size_t)(n0 + r) * Kt + k0 + c);
    }
    __syncthreads();
    compute_stage(0);
  }
#endif

  // Epilogue: h = silu(g) * u, f16
#pragma unroll
  for (int mt = 0; mt < 2; ++mt)
#pragma unroll
    for (int nt = 0; nt < 2; ++nt)
#pragma unroll
      for (int v = 0; v < 8; ++v) {
        float g = accG[mt][nt][v];
        float u = accU[mt][nt][v];
        float sg = g / (1.0f + __expf(-g));
        int m = m0 + wm + mt * 16 + half * 8 + v;
        int n = n0 + wn + nt * 16 + row;
        H[(size_t)m * Nt + n] = (_Float16)(sg * u);
      }
}

// ---------------------------------------------------------------------------
// Down GEMM: out[M][D_MODEL] = H @ Wd, f32 out.
// BM=256 x BN=128, 1024 threads = 32 wave32 (8x4 grid, 32x32 wave tiles),
// KB=64 double-buffered async pipeline (~110KB LDS).
// ---------------------------------------------------------------------------
#define DBM 256
#define DBN 128

__global__ __launch_bounds__(1024) void awq_gemm_down(const _Float16* __restrict__ A,
                                                      const _Float16* __restrict__ B,
                                                      float* __restrict__ O) {
  const int Kt = D_FF, Nt = D_MODEL;
  __shared__ __align__(16) _Float16 As[2][DBM][KP];
  __shared__ __align__(16) _Float16 Bs[2][DBN][KP];

  const int t    = threadIdx.x;
  const int m0   = blockIdx.x * DBM;
  const int n0   = blockIdx.y * DBN;
  const int lane = t & 31, w = t >> 5;
  const int wm   = (w >> 2) * 32;
  const int wn   = (w & 3) * 32;
  const int half = lane >> 4, row = lane & 15;

  v8f acc[2][2] = {};

  auto compute_stage = [&](int b) {
#pragma unroll
    for (int ks = 0; ks < 2; ++ks) {
      const int kb = ks * 32;
      v16h af[2];
#pragma unroll
      for (int mt = 0; mt < 2; ++mt)
        af[mt] = load_a_frag(&As[b][wm + mt * 16 + row][0], kb, half);
#pragma unroll
      for (int nt = 0; nt < 2; ++nt) {
        const v16h bf = load_b_frag(&Bs[b][wn + nt * 16 + row][0], kb, half);
#pragma unroll
        for (int mt = 0; mt < 2; ++mt)
          acc[mt][nt] = __builtin_amdgcn_wmma_f32_16x16x32_f16(
              false, af[mt], false, bf, (short)0, acc[mt][nt], false, false);
      }
    }
  };

#if HAVE_ASYNC_LDS
  auto issue_stage = [&](int k0, int b) {
#pragma unroll
    for (int i = 0; i < 2; ++i) {        // A tile: 2048 x 16B
      int idx = t + i * 1024;
      int r = idx >> 3, c = (idx & 7) * 8;
      async_copy16(A + (size_t)(m0 + r) * Kt + k0 + c, &As[b][r][c]);
    }
    {                                    // B tile: 1024 x 16B
      int r = t >> 3, c = (t & 7) * 8;
      async_copy16(B + (size_t)(n0 + r) * Kt + k0 + c, &Bs[b][r][c]);
    }
  };
  issue_stage(0, 0);
  for (int k0 = 0; k0 < Kt; k0 += 2 * KB) {  // Kt/KB = 172 stages (even)
    wait_async0();
    __syncthreads();
    if (k0 + KB < Kt) issue_stage(k0 + KB, 1);
    compute_stage(0);
    wait_async0();
    __syncthreads();
    if (k0 + 2 * KB < Kt) issue_stage(k0 + 2 * KB, 0);
    compute_stage(1);
  }
#else
  for (int k0 = 0; k0 < Kt; k0 += KB) {
    __syncthreads();
#pragma unroll
    for (int i = 0; i < 2; ++i) {
      int idx = t + i * 1024;
      int r = idx >> 3, c = (idx & 7) * 8;
      *(uint4*)&As[0][r][c] = *(const uint4*)(A + (size_t)(m0 + r) * Kt + k0 + c);
    }
    {
      int r = t >> 3, c = (t & 7) * 8;
      *(uint4*)&Bs[0][r][c] = *(const uint4*)(B + (size_t)(n0 + r) * Kt + k0 + c);
    }
    __syncthreads();
    compute_stage(0);
  }
#endif

#pragma unroll
  for (int mt = 0; mt < 2; ++mt)
#pragma unroll
    for (int nt = 0; nt < 2; ++nt)
#pragma unroll
      for (int v = 0; v < 8; ++v) {
        int m = m0 + wm + mt * 16 + half * 8 + v;
        int n = n0 + wn + nt * 16 + row;
        O[(size_t)m * Nt + n] = acc[mt][nt][v];
      }
}

// ---------------------------------------------------------------------------
// Host-side orchestration.  Workspace layout (bytes):
//   xh @ 0 (64MB) | wg @ 67108864 | wu @ 157286400 | wd @ 247463936
//   hh @ 337641472 ... total 517,996,544
// ---------------------------------------------------------------------------
extern "C" void kernel_launch(void* const* d_in, const int* in_sizes, int n_in,
                              void* d_out, int out_size, void* d_ws, size_t ws_size,
                              hipStream_t stream) {
  const float* x      = (const float*)d_in[0];
  const int*   q_gate = (const int*)d_in[1];
  const int*   z_gate = (const int*)d_in[2];
  const float* s_gate = (const float*)d_in[3];
  const int*   q_up   = (const int*)d_in[4];
  const int*   z_up   = (const int*)d_in[5];
  const float* s_up   = (const float*)d_in[6];
  const int*   q_down = (const int*)d_in[7];
  const int*   z_down = (const int*)d_in[8];
  const float* s_down = (const float*)d_in[9];
  float* out = (float*)d_out;

  char* ws = (char*)d_ws;
  _Float16* xh = (_Float16*)(ws);
  _Float16* wg = (_Float16*)(ws + 67108864ull);
  _Float16* wu = (_Float16*)(ws + 157286400ull);
  _Float16* wd = (_Float16*)(ws + 247463936ull);
  _Float16* hh = (_Float16*)(ws + 337641472ull);

  awq_x_to_h<<<(M_DIM * D_MODEL / 4 + 255) / 256, 256, 0, stream>>>(
      x, xh, M_DIM * D_MODEL / 4);

  awq_dequant_t<<<dim3(D_FF / 64, D_MODEL / 64), 256, 0, stream>>>(
      q_gate, z_gate, s_gate, wg, D_MODEL, D_FF);
  awq_dequant_t<<<dim3(D_FF / 64, D_MODEL / 64), 256, 0, stream>>>(
      q_up, z_up, s_up, wu, D_MODEL, D_FF);
  awq_dequant_t<<<dim3(D_MODEL / 64, D_FF / 64), 256, 0, stream>>>(
      q_down, z_down, s_down, wd, D_FF, D_MODEL);

  awq_gemm_gateup<<<dim3(M_DIM / GBM, D_FF / GBN), 1024, 0, stream>>>(xh, wg, wu, hh);
  awq_gemm_down<<<dim3(M_DIM / DBM, D_MODEL / DBN), 1024, 0, stream>>>(hh, wd, out);
}
